// Attention_33432025432597
// MI455X (gfx1250) — matrix operand order repaired and checked by reading, not compiled, not run on hardware
//
#include <hip/hip_runtime.h>
#include <hip/hip_bf16.h>

typedef __bf16 bf16;
typedef __attribute__((ext_vector_type(16))) __bf16 v16bf;
typedef __attribute__((ext_vector_type(4)))  __bf16 v4bf;
typedef __attribute__((ext_vector_type(8)))  float  v8f;
typedef __attribute__((ext_vector_type(4)))  float  f32x4;
typedef unsigned int u32;
typedef __attribute__((ext_vector_type(4))) unsigned int u32x4;
typedef __attribute__((ext_vector_type(8))) int i32x8;
typedef __attribute__((ext_vector_type(4))) int i32x4;

#define N_EMBED 1024
#define SEQ     2048
#define NB      4
#define M_TOTAL (NB * SEQ)   // 8192

// ---------------- WMMA fragment helpers (CDNA5 wave32 layouts) ----------------

__device__ __forceinline__ v16bf load_a_frag(const bf16* base, int ld) {
  const int lane = threadIdx.x & 31;
  const int r = lane & 15, hi = lane >> 4;
  v16bf a;
#pragma unroll
  for (int v = 0; v < 8; ++v) {
    const int k = hi * 16 + 2 * v;
    a[2 * v]     = base[r * ld + k];
    a[2 * v + 1] = base[r * ld + k + 1];
  }
  return a;
}

// B 32x16 bf16 (KxN), LDS tile stored N-major: tile[n][k], ld = K elems.
__device__ __forceinline__ v16bf load_b_frag_nk(const bf16* base, int ld) {
  const int lane = threadIdx.x & 31;
  const int n = lane & 15, hi = lane >> 4;
  v16bf b;
#pragma unroll
  for (int v = 0; v < 8; ++v) {
    const int k = hi * 16 + 2 * v;
    b[2 * v]     = base[n * ld + k];
    b[2 * v + 1] = base[n * ld + k + 1];
  }
  return b;
}

// B fragment from K-major LDS tile: tile[k][n], ld = N elems.
__device__ __forceinline__ v16bf load_b_frag_kn(const bf16* base, int ld) {
  const int lane = threadIdx.x & 31;
  const int n = lane & 15, hi = lane >> 4;
  v16bf b;
#pragma unroll
  for (int v = 0; v < 8; ++v) {
    const int k = hi * 16 + 2 * v;
    b[2 * v]     = base[k * ld + n];
    b[2 * v + 1] = base[(k + 1) * ld + n];
  }
  return b;
}

__device__ __forceinline__ v8f wmma_bf16(v16bf a, v16bf b, v8f c) {
  return __builtin_amdgcn_wmma_f32_16x16x32_bf16(false, a, false, b,
                                                 (short)0, c, false, false);
}

// -------- Tensor Data Mover: 2-D bf16 tile load (D# built per ISA 08 §8) -----
__device__ __forceinline__ void tdm_load_2d(u32 lds_addr, const void* gptr,
                                            u32 tensor_d0, u32 tensor_d1,
                                            u32 tile_d0, u32 tile_d1,
                                            u32 stride0) {
  unsigned long long ga = (unsigned long long)gptr;
  u32x4 g0;
  g0[0] = 1u;                                   // count=1 (valid), user mode
  g0[1] = lds_addr;                             // lds_addr [63:32]
  g0[2] = (u32)(ga & 0xFFFFFFFFu);              // global_addr [95:64]
  g0[3] = (u32)((ga >> 32) & 0x01FFFFFFu)       // global_addr [120:96]
          | (2u << 30);                         // type=2 ("image") [127:126]
  i32x8 g1;
  g1[0] = (int)(1u << 16);                      // wg_mask=0, data_size=1 (2B)
  g1[1] = (int)((tensor_d0 & 0xFFFFu) << 16);   // tensor_dim0 [79:48]
  g1[2] = (int)(((tensor_d0 >> 16) & 0xFFFFu) |
                ((tensor_d1 & 0xFFFFu) << 16)); // tensor_dim1 [111:80]
  g1[3] = (int)(((tensor_d1 >> 16) & 0xFFFFu) |
                ((tile_d0 & 0xFFFFu) << 16));   // tile_dim0 [127:112]
  g1[4] = (int)(tile_d1 & 0xFFFFu);             // tile_dim1; tile_dim2=0
  g1[5] = (int)stride0;                         // tensor_dim0_stride [207:160]
  g1[6] = 0;
  g1[7] = 0;
  i32x4 z4 = {0, 0, 0, 0};
  i32x8 z8 = {0, 0, 0, 0, 0, 0, 0, 0};
  __builtin_amdgcn_tensor_load_to_lds(g0, g1, z4, z4, z8, 0);
}

__device__ __forceinline__ u32 lds_off(const void* p) {
  return (u32)(unsigned long long)p;            // low 32 bits = LDS byte offset
}

// ---------------- Kernel 1: QKV projection  q = x @ W^T + b  (bf16 out) -------
// 128x128 tile/WG, 8 waves, double-buffered LDS: stage tile i+1 while
// computing tile i (single barrier per iteration).
__global__ __launch_bounds__(256) void qkv_kernel(
    const float* __restrict__ x,
    const float* __restrict__ Wq, const float* __restrict__ Wk,
    const float* __restrict__ Wv,
    const float* __restrict__ bq, const float* __restrict__ bk,
    const float* __restrict__ bv,
    bf16* __restrict__ qo, bf16* __restrict__ ko, bf16* __restrict__ vo) {
  const float* W    = (blockIdx.z == 0) ? Wq : (blockIdx.z == 1) ? Wk : Wv;
  const float* bias = (blockIdx.z == 0) ? bq : (blockIdx.z == 1) ? bk : bv;
  bf16*        out  = (blockIdx.z == 0) ? qo : (blockIdx.z == 1) ? ko : vo;

  __shared__ __align__(16) bf16 As[2][128 * 32];
  __shared__ __align__(16) bf16 Bs[2][128 * 32];

  const int tid = threadIdx.x;
  const int m0 = blockIdx.y * 128;
  const int n0 = blockIdx.x * 128;
  const int w  = tid >> 5;
  const int wm = (w >> 1) * 32;
  const int wn = (w & 1) * 64;
  const int lane = tid & 31, r = lane & 15, hi = lane >> 4;

  float bj[4];
#pragma unroll
  for (int j = 0; j < 4; ++j) bj[j] = bias[n0 + wn + 16 * j + r];

  // Stage one 128x32 f32 tile pair (x, W) -> bf16 LDS buffer `bb`.
  auto stage = [&](int d0, int bb) {
#pragma unroll
    for (int i = 0; i < 4; ++i) {
      int idx = tid + i * 256;
      int row = idx >> 3, c4 = (idx & 7) * 4;
      f32x4 fa = *(const f32x4*)(x + (size_t)(m0 + row) * N_EMBED + d0 + c4);
      f32x4 fb = *(const f32x4*)(W + (size_t)(n0 + row) * N_EMBED + d0 + c4);
      v4bf ha, hb;
#pragma unroll
      for (int t = 0; t < 4; ++t) { ha[t] = (bf16)fa[t]; hb[t] = (bf16)fb[t]; }
      *(v4bf*)&As[bb][row * 32 + c4] = ha;
      *(v4bf*)&Bs[bb][row * 32 + c4] = hb;
    }
  };

  v8f c[2][4] = {};
  stage(0, 0);

  for (int it = 0; it < N_EMBED / 32; ++it) {
    const int cur = it & 1;
    __syncthreads();                       // stage(it) visible to all waves
    if (it + 1 < N_EMBED / 32) stage((it + 1) * 32, cur ^ 1);

    v16bf a[2], b[4];
#pragma unroll
    for (int i = 0; i < 2; ++i)
      a[i] = load_a_frag(&As[cur][(wm + 16 * i) * 32], 32);
#pragma unroll
    for (int j = 0; j < 4; ++j)
      b[j] = load_b_frag_nk(&Bs[cur][(wn + 16 * j) * 32], 32);
#pragma unroll
    for (int i = 0; i < 2; ++i)
#pragma unroll
      for (int j = 0; j < 4; ++j)
        c[i][j] = wmma_bf16(a[i], b[j], c[i][j]);
  }

#pragma unroll
  for (int i = 0; i < 2; ++i)
#pragma unroll
    for (int j = 0; j < 4; ++j)
#pragma unroll
      for (int v = 0; v < 8; ++v) {
        int m = m0 + wm + 16 * i + v + 8 * hi;
        int n = n0 + wn + 16 * j + r;
        out[(size_t)m * N_EMBED + n] = (bf16)(c[i][j][v] + bj[j]);
      }
}

// -------- Kernel 2: fused scores + softmax  P = softmax(Q K^T / sqrt(d)) -----
// One WG per (batch, 16 query rows). 16x2048 f32 panel in LDS. K chunks are
// TDM-DMA'd into ping-pong buffers: wave 0 issues chunk i+1, waits
// s_wait_tensorcnt<=1 (in-order => chunk i resident), barrier releases waves.
__global__ __launch_bounds__(256) void scores_softmax_kernel(
    const bf16* __restrict__ Q, const bf16* __restrict__ K,
    bf16* __restrict__ P) {
  extern __shared__ __align__(16) char smem[];
  float* sc  = (float*)smem;                          // 131072 B
  bf16*  Qs  = (bf16*)(smem + 131072);                //  32768 B
  bf16*  Ks0 = (bf16*)(smem + 131072 + 32768);        //  32768 B
  bf16*  Ks1 = (bf16*)(smem + 131072 + 65536);        //  32768 B
  float* red = (float*)(smem + 131072 + 98304);       //   1024 B

  const int tid = threadIdx.x;
  const int b   = blockIdx.y;
  const int m0  = blockIdx.x * 16;
  const bf16* Qb = Q + (size_t)(b * SEQ + m0) * N_EMBED;
  const bf16* Kb = K + (size_t)b * SEQ * N_EMBED;

  const int w    = tid >> 5;
  const int lane = tid & 31, nn = lane & 15, hi = lane >> 4;
  const float scale = 0.03125f;  // 1/sqrt(1024)
  const int NCHUNK = 128;        // 4 key superblocks x 32 d-steps

  // chunk i covers keys [(i>>5)*512, +512) x d [(i&31)*32, +32)
  auto kaddr = [&](int i) -> const char* {
    return (const char*)Kb +
           ((size_t)((i >> 5) * 512) * N_EMBED + (size_t)(i & 31) * 32) * 2;
  };

  if (w == 0) {
    tdm_load_2d(lds_off(Qs), Qb, N_EMBED, SEQ, N_EMBED, 16, N_EMBED);
    tdm_load_2d(lds_off(Ks0), kaddr(0), N_EMBED, SEQ, 32, 512, N_EMBED);
  }

  for (int kb = 0; kb < 4; ++kb) {
    const int keybase = kb * 512;
    v8f c[4] = {};
    for (int dk = 0; dk < 32; ++dk) {
      const int it = kb * 32 + dk;
      bf16* Kcur = (it & 1) ? Ks1 : Ks0;
      if (w == 0) {
        if (it + 1 < NCHUNK) {
          tdm_load_2d(lds_off((it & 1) ? Ks0 : Ks1), kaddr(it + 1),
                      N_EMBED, SEQ, 32, 512, N_EMBED);
          __builtin_amdgcn_s_wait_tensorcnt(1);  // chunk it (and Q) resident
        } else {
          __builtin_amdgcn_s_wait_tensorcnt(0);
        }
      }
      __syncthreads();  // release consumers for Kcur

      v16bf a = load_a_frag(&Qs[dk * 32], N_EMBED);
#pragma unroll
      for (int j = 0; j < 4; ++j) {
        v16bf bf = load_b_frag_nk(&Kcur[(w * 64 + 16 * j) * 32], 32);
        c[j] = wmma_bf16(a, bf, c[j]);
      }
      __syncthreads();  // all waves done with Kcur before it is overwritten
    }
#pragma unroll
    for (int j = 0; j < 4; ++j)
#pragma unroll
      for (int v = 0; v < 8; ++v) {
        int m   = v + 8 * hi;
        int key = keybase + w * 64 + 16 * j + nn;
        sc[m * SEQ + key] = c[j][v] * scale;
      }
  }
  __syncthreads();

  // Row softmax: 16 threads per row, strided columns.
  const int r = tid >> 4, s = tid & 15;
  float mx = -3.4e38f;
  for (int i = 0; i < SEQ / 16; ++i) mx = fmaxf(mx, sc[r * SEQ + s + i * 16]);
  red[r * 16 + s] = mx;
  __syncthreads();
  mx = red[r * 16];
#pragma unroll
  for (int i = 1; i < 16; ++i) mx = fmaxf(mx, red[r * 16 + i]);

  float sum = 0.f;
  for (int i = 0; i < SEQ / 16; ++i) {
    float e = __expf(sc[r * SEQ + s + i * 16] - mx);
    sc[r * SEQ + s + i * 16] = e;
    sum += e;
  }
  __syncthreads();
  red[r * 16 + s] = sum;
  __syncthreads();
  sum = 0.f;
#pragma unroll
  for (int i = 0; i < 16; ++i) sum += red[r * 16 + i];
  const float inv = 1.0f / sum;

  bf16* Pb = P + (size_t)(b * SEQ + m0) * SEQ;
  for (int i = 0; i < SEQ / 16; ++i) {
    int col = s + i * 16;
    Pb[(size_t)r * SEQ + col] = (bf16)(sc[r * SEQ + col] * inv);
  }
}

// ---------------- Kernel 3: attn = P @ V, out = embedded + attn --------------
// P and V tiles TDM-DMA'd into ping-pong LDS buffers, pipelined as above
// (two tiles in flight => wait tensorcnt<=2 keeps the current pair resident).
__global__ __launch_bounds__(256) void pv_kernel(
    const bf16* __restrict__ P, const bf16* __restrict__ V,
    const float* __restrict__ emb, float* __restrict__ out) {
  __shared__ __align__(16) bf16 As[2][128 * 32];  // P tile [m][k]
  __shared__ __align__(16) bf16 Vs[2][32 * 128];  // V tile [k][n]

  const int tid = threadIdx.x;
  const int b  = blockIdx.z;
  const int m0 = blockIdx.y * 128;
  const int n0 = blockIdx.x * 128;
  const bf16* Pb = P + (size_t)b * SEQ * SEQ;
  const bf16* Vb = V + (size_t)b * SEQ * N_EMBED;
  const int w  = tid >> 5;
  const int wm = (w >> 1) * 32;
  const int wn = (w & 1) * 64;
  const int NIT = SEQ / 32;  // 64

  auto issue = [&](int it, int bb) {
    const char* gp =
        (const char*)Pb + ((size_t)m0 * SEQ + (size_t)it * 32) * 2;
    const char* gv =
        (const char*)Vb + ((size_t)it * 32 * N_EMBED + n0) * 2;
    tdm_load_2d(lds_off(As[bb]), gp, SEQ, SEQ, 32, 128, SEQ);
    tdm_load_2d(lds_off(Vs[bb]), gv, N_EMBED, SEQ, 128, 32, N_EMBED);
  };

  v8f c[2][4] = {};
  if (w == 0) issue(0, 0);

  for (int it = 0; it < NIT; ++it) {
    const int cur = it & 1;
    if (w == 0) {
      if (it + 1 < NIT) {
        issue(it + 1, cur ^ 1);
        __builtin_amdgcn_s_wait_tensorcnt(2);  // pair `it` resident
      } else {
        __builtin_amdgcn_s_wait_tensorcnt(0);
      }
    }
    __syncthreads();

    v16bf a[2], bf[4];
#pragma unroll
    for (int i = 0; i < 2; ++i)
      a[i] = load_a_frag(&As[cur][(wm + 16 * i) * 32], 32);
#pragma unroll
    for (int j = 0; j < 4; ++j)
      bf[j] = load_b_frag_kn(&Vs[cur][wn + 16 * j], 128);
#pragma unroll
    for (int i = 0; i < 2; ++i)
#pragma unroll
      for (int j = 0; j < 4; ++j)
        c[i][j] = wmma_bf16(a[i], bf[j], c[i][j]);
    __syncthreads();  // done with `cur` before wave 0 refills it
  }

  const int lane = tid & 31, r = lane & 15, hi = lane >> 4;
  const float* eb = emb + (size_t)b * SEQ * N_EMBED;
  float*       ob = out + (size_t)b * SEQ * N_EMBED;
#pragma unroll
  for (int i = 0; i < 2; ++i)
#pragma unroll
    for (int j = 0; j < 4; ++j)
#pragma unroll
      for (int v = 0; v < 8; ++v) {
        int m = m0 + wm + 16 * i + v + 8 * hi;
        int n = n0 + wn + 16 * j + r;
        ob[(size_t)m * N_EMBED + n] = eb[(size_t)m * N_EMBED + n] + c[i][j][v];
      }
}

// -----------------------------------------------------------------------------
extern "C" void kernel_launch(void* const* d_in, const int* in_sizes, int n_in,
                              void* d_out, int out_size, void* d_ws,
                              size_t ws_size, hipStream_t stream) {
  const float* emb = (const float*)d_in[0];
  const float* Wq  = (const float*)d_in[1];
  const float* bq  = (const float*)d_in[2];
  const float* Wk  = (const float*)d_in[3];
  const float* bk  = (const float*)d_in[4];
  const float* Wv  = (const float*)d_in[5];
  const float* bv  = (const float*)d_in[6];
  float* out = (float*)d_out;

  char* ws = (char*)d_ws;
  bf16* Qw = (bf16*)(ws);                         // 16 MB
  bf16* Kw = (bf16*)(ws + ((size_t)16 << 20));    // 16 MB
  bf16* Vw = (bf16*)(ws + ((size_t)32 << 20));    // 16 MB
  bf16* Pw = (bf16*)(ws + ((size_t)48 << 20));    // 32 MB

  dim3 g1(N_EMBED / 128, M_TOTAL / 128, 3);
  qkv_kernel<<<g1, 256, 0, stream>>>(emb, Wq, Wk, Wv, bq, bk, bv, Qw, Kw, Vw);

  dim3 g2(SEQ / 16, NB);
  size_t smem2 = 131072 + 32768 + 2 * 32768 + 1024;  // 230400 B (<320KB/WG)
  scores_softmax_kernel<<<g2, 256, smem2, stream>>>(Qw, Kw, Pw);

  dim3 g3(N_EMBED / 128, SEQ / 128, NB);
  pv_kernel<<<g3, 256, 0, stream>>>(Pw, Vw, emb, out);
}
